// DiscoveryEngineModel_70437463654722
// MI455X (gfx1250) — matrix operand order repaired and checked by reading, compiled, not run on hardware
//
#include <hip/hip_runtime.h>

// ===========================================================================
// EGNN layer for MI455X (gfx1250): bf16 WMMA 16x16x32, edges tiled 16/wave.
// ===========================================================================

typedef __attribute__((ext_vector_type(16))) __bf16 v16bf;
typedef __attribute__((ext_vector_type(8)))  float  v8f;

#define NW      8      // waves per 256-thread block (wave32)
#define EBLOCKS 1250   // edge kernel grid: 1250*8*16 = 160000 edges/sweep -> 10 sweeps
#define NBLOCKS 625    // node kernel grid

// ---------------- bf16 helpers (native converts -> v_cvt_pk_bf16_f32) ------
__device__ __forceinline__ unsigned short f2bf(float f) {
  return __builtin_bit_cast(unsigned short, (__bf16)f);
}
__device__ __forceinline__ unsigned int pack2(float a, float b) {
  unsigned short ua = __builtin_bit_cast(unsigned short, (__bf16)a);
  unsigned short ub = __builtin_bit_cast(unsigned short, (__bf16)b);
  return (unsigned int)ua | ((unsigned int)ub << 16);
}
// softplus on hardware transcendentals (v_exp_f32 / v_log_f32); these
// co-execute with XDL/WMMA ops per the CDNA5 scheduling rules.
__device__ __forceinline__ float softplus_f(float v) {
  return fmaxf(v, 0.0f) + __logf(1.0f + __expf(-fabsf(v)));
}

union Frag {
  v16bf v;
  unsigned short s[16];
  unsigned int   u[8];
  uint4          q[2];
};

__device__ __forceinline__ v8f v8f_zero() {
  v8f z = {0.f,0.f,0.f,0.f,0.f,0.f,0.f,0.f};
  return z;
}

__device__ __forceinline__ v8f wmma_bf16(v16bf a, v16bf b, v8f c) {
  // (neg_a, A, neg_b, B, c_mod, C, reuse_a, reuse_b)
  return __builtin_amdgcn_wmma_f32_16x16x32_bf16(false, a, false, b, (short)0, c,
                                                 false, false);
}

// B fragment (32x16 bf16). Weights stored transposed: row n, K contiguous,
// row stride rs (ushorts). elems j<8 -> k = kbase+hi8+j ; j>=8 -> k = kbase+16+hi8+(j-8).
__device__ __forceinline__ v16bf bfrag(const unsigned short* wt, int rs, int n,
                                       int kbase, int khi16) {
  Frag f;
  const unsigned short* p = wt + n * rs + kbase + (khi16 >> 1);
  f.q[0] = *(const uint4*)(p);
  f.q[1] = *(const uint4*)(p + 16);
  return f.v;
}

// A fragment (16x32 bf16) from LDS stage, row-major [m][K], stride rs.
// elems j<8 -> k = kbase+hi8+j ; j>=8 -> k = kbase+16+hi8+(j-8)
__device__ __forceinline__ v16bf afrag_lds(const unsigned short* st, int rs, int m,
                                           int kbase, int hi8) {
  Frag f;
  const unsigned short* p = st + m * rs + kbase + hi8;
  f.q[0] = *(const uint4*)(p);
  f.q[1] = *(const uint4*)(p + 16);
  return f.v;
}

// A fragment from a global f32 feature row (64 contiguous floats), scaled,
// converted to bf16 on the fly.
__device__ __forceinline__ v16bf afrag_g(const float* row, int kbase, int hi8,
                                         float scale) {
  const float* p = row + kbase + hi8;
  float4 a = *(const float4*)(p);
  float4 b = *(const float4*)(p + 4);
  float4 c = *(const float4*)(p + 16);
  float4 d = *(const float4*)(p + 20);
  Frag f;
  f.u[0] = pack2(a.x * scale, a.y * scale);
  f.u[1] = pack2(a.z * scale, a.w * scale);
  f.u[2] = pack2(b.x * scale, b.y * scale);
  f.u[3] = pack2(b.z * scale, b.w * scale);
  f.u[4] = pack2(c.x * scale, c.y * scale);
  f.u[5] = pack2(c.z * scale, c.w * scale);
  f.u[6] = pack2(d.x * scale, d.y * scale);
  f.u[7] = pack2(d.z * scale, d.w * scale);
  return f.v;
}

// ---------------- workspace zero ----------------
__global__ void zero_f32(float* __restrict__ p, size_t n) {
  size_t i = (size_t)blockIdx.x * blockDim.x + threadIdx.x;
  size_t stride = (size_t)gridDim.x * blockDim.x;
  for (; i < n; i += stride) p[i] = 0.0f;
}

// ---------------- weight prep: f32 -> transposed/padded bf16 ----------------
// gWE1: fused [128 out][168 k] (out 0..63 = phi_e L1, 64..127 = phi_v L1; k<130 real)
// gWE2/gWE3/gWH2: [64][72]; gWV2: [16][72] (col0 real); gWH1: [64][168] (k<129 real)
// gBias: [0..127]=bE1|bV1, [128..191]=bE2, [192..255]=bE3, [256]=bV2,
//        [257..320]=bH1, [321..384]=bH2
__global__ void prep_weights(
    const float* __restrict__ We1, const float* __restrict__ Wv1,
    const float* __restrict__ We2, const float* __restrict__ We3,
    const float* __restrict__ Wv2, const float* __restrict__ Wh1,
    const float* __restrict__ Wh2,
    const float* __restrict__ be1, const float* __restrict__ bv1,
    const float* __restrict__ be2, const float* __restrict__ be3,
    const float* __restrict__ bv2, const float* __restrict__ bh1,
    const float* __restrict__ bh2,
    unsigned short* __restrict__ gWE1, unsigned short* __restrict__ gWE2,
    unsigned short* __restrict__ gWE3, unsigned short* __restrict__ gWV2,
    unsigned short* __restrict__ gWH1, unsigned short* __restrict__ gWH2,
    float* __restrict__ gBias) {
  int tid = blockIdx.x * blockDim.x + threadIdx.x;
  int stride = gridDim.x * blockDim.x;
  for (int idx = tid; idx < 128 * 168; idx += stride) {
    int n = idx / 168, k = idx % 168;
    float w = 0.0f;
    if (k < 130) w = (n < 64) ? We1[k * 64 + n] : Wv1[k * 64 + (n - 64)];
    gWE1[idx] = f2bf(w);
  }
  for (int idx = tid; idx < 64 * 72; idx += stride) {
    int n = idx / 72, k = idx % 72;
    gWE2[idx] = f2bf(k < 64 ? We2[k * 64 + n] : 0.0f);
    gWE3[idx] = f2bf(k < 64 ? We3[k * 64 + n] : 0.0f);
    gWH2[idx] = f2bf(k < 64 ? Wh2[k * 64 + n] : 0.0f);
  }
  for (int idx = tid; idx < 16 * 72; idx += stride) {
    int n = idx / 72, k = idx % 72;
    gWV2[idx] = f2bf((n == 0 && k < 64) ? Wv2[k] : 0.0f);
  }
  for (int idx = tid; idx < 64 * 168; idx += stride) {
    int n = idx / 168, k = idx % 168;
    gWH1[idx] = f2bf(k < 129 ? Wh1[k * 64 + n] : 0.0f);
  }
  if (tid < 64) {
    gBias[tid]       = be1[tid];
    gBias[64 + tid]  = bv1[tid];
    gBias[128 + tid] = be2[tid];
    gBias[192 + tid] = be3[tid];
    gBias[257 + tid] = bh1[tid];
    gBias[321 + tid] = bh2[tid];
  }
  if (tid == 0) gBias[256] = bv2[0];
}

// ---------------- edge kernel ----------------
// Each wave processes a tile of 16 edges:
//  L1 fused: [16 x 160] @ [160 x 128]  (40 WMMA)   A built from global gathers
//  phi_v L2: [16 x 64] @ [64 x 16pad]  (2 WMMA)  -> v_w ; scatter m_v, cnt
//  phi_e L2: [16 x 64] @ [64 x 64]     (8 WMMA)
//  phi_e L3: [16 x 64] @ [64 x 64]     (8 WMMA)  -> atomic scatter m_h
__global__ __launch_bounds__(256) void egnn_edge_kernel(
    const float* __restrict__ x, const float* __restrict__ pos,
    const float* __restrict__ vel, const int* __restrict__ srcIdx,
    const int* __restrict__ dstIdx,
    const unsigned short* __restrict__ gWE1, const unsigned short* __restrict__ gWE2,
    const unsigned short* __restrict__ gWE3, const unsigned short* __restrict__ gWV2,
    const float* __restrict__ gBias,
    float* __restrict__ mh, float* __restrict__ mv, float* __restrict__ cnt,
    int E, int ntiles, int niters) {
  __shared__ __attribute__((aligned(16))) unsigned short sWE1[128 * 168];
  __shared__ __attribute__((aligned(16))) unsigned short sWE2[64 * 72];
  __shared__ __attribute__((aligned(16))) unsigned short sWE3[64 * 72];
  __shared__ __attribute__((aligned(16))) unsigned short sWV2[16 * 72];
  __shared__ __attribute__((aligned(16))) unsigned short sStage[NW][16][136];
  __shared__ float sBE1[128], sBE2[64], sBE3[64];
  __shared__ float sBV2;
  __shared__ int   sSrc[NW][16], sDst[NW][16];
  __shared__ float sRelX[NW][16], sRelY[NW][16], sDist[NW][16], sDot[NW][16];
  __shared__ float sVW[NW][16];

  // ---- cooperative weight load into LDS (amortized over many tiles) ----
  for (int i = threadIdx.x; i < 128 * 168 / 2; i += blockDim.x)
    ((unsigned int*)sWE1)[i] = ((const unsigned int*)gWE1)[i];
  for (int i = threadIdx.x; i < 64 * 72 / 2; i += blockDim.x) {
    ((unsigned int*)sWE2)[i] = ((const unsigned int*)gWE2)[i];
    ((unsigned int*)sWE3)[i] = ((const unsigned int*)gWE3)[i];
  }
  for (int i = threadIdx.x; i < 16 * 72 / 2; i += blockDim.x)
    ((unsigned int*)sWV2)[i] = ((const unsigned int*)gWV2)[i];
  for (int i = threadIdx.x; i < 128; i += blockDim.x) sBE1[i] = gBias[i];
  for (int i = threadIdx.x; i < 64; i += blockDim.x) {
    sBE2[i] = gBias[128 + i];
    sBE3[i] = gBias[192 + i];
  }
  if (threadIdx.x == 0) sBV2 = gBias[256];
  __syncthreads();

  const int wid   = threadIdx.x >> 5;
  const int lane  = threadIdx.x & 31;
  const int m     = lane & 15;
  const int hi8   = (lane >> 4) << 3;  // 0 | 8  (A k-offset, D row-offset)
  const int khi16 = hi8 << 1;          // 0 | 16 (bfrag halves it: B k-offset 0 | 8)

  for (int it = 0; it < niters; ++it) {
    int tile = (it * gridDim.x + blockIdx.x) * NW + wid;
    bool active = tile < ntiles;

    // ---- per-edge geometry staging ----
    if (active && lane < 16) {
      int e = tile * 16 + lane;
      bool ev = e < E;
      int ec = ev ? e : 0;
      int s = srcIdx[ec], d = dstIdx[ec];
      sSrc[wid][lane] = s;
      sDst[wid][lane] = ev ? d : -1;
      float rx = pos[2 * s]     - pos[2 * d];
      float ry = pos[2 * s + 1] - pos[2 * d + 1];
      float vx = vel[2 * s]     - vel[2 * d];
      float vy = vel[2 * s + 1] - vel[2 * d + 1];
      sRelX[wid][lane] = rx;
      sRelY[wid][lane] = ry;
      sDist[wid][lane] = rx * rx + ry * ry;
      sDot[wid][lane]  = vx * rx + vy * ry;
    }
    __syncthreads();

    // ---- layer 1 fused (phi_e L1 + phi_v L1): tmp[16x160] @ W[160x128] ----
    v8f acc[8];
#pragma unroll
    for (int i = 0; i < 8; ++i) acc[i] = v8f_zero();
    if (active) {
      int dd = sDst[wid][m]; if (dd < 0) dd = 0;
      int ss = sSrc[wid][m];
      const float* xi = x + (size_t)dd * 64;  // x_i = x[dst]
      const float* xj = x + (size_t)ss * 64;  // x_j = x[src]
#pragma unroll
      for (int ks = 0; ks < 4; ++ks) {
        v16bf a = (ks < 2) ? afrag_g(xi, ks * 32, hi8, 1.0f)
                           : afrag_g(xj, (ks - 2) * 32, hi8, 1.0f);
#pragma unroll
        for (int nt = 0; nt < 8; ++nt)
          acc[nt] = wmma_bf16(a, bfrag(sWE1, 168, nt * 16 + m, ks * 32, khi16),
                              acc[nt]);
      }
      {  // k-step 4: features 128,129 = dist_sq, dot_vr (rest zero-weighted)
        Frag fa;
#pragma unroll
        for (int i = 0; i < 8; ++i) fa.u[i] = 0u;
        if (lane < 16) fa.u[0] = pack2(sDist[wid][m], sDot[wid][m]);
#pragma unroll
        for (int nt = 0; nt < 8; ++nt)
          acc[nt] = wmma_bf16(fa.v, bfrag(sWE1, 168, nt * 16 + m, 128, khi16),
                              acc[nt]);
      }
      // bias + softplus, stage transposed (rows = edges) for next layers
#pragma unroll
      for (int nt = 0; nt < 8; ++nt) {
        int n = nt * 16 + m;
        float bias = sBE1[n];
#pragma unroll
        for (int r = 0; r < 8; ++r)
          sStage[wid][r + hi8][n] = f2bf(softplus_f(acc[nt][r] + bias));
      }
    }
    __syncthreads();

    // ---- phi_v layer 2 (padded to 16 outputs; col 0 real) ----
    if (active) {
      v8f av = v8f_zero();
#pragma unroll
      for (int ks = 0; ks < 2; ++ks)
        av = wmma_bf16(afrag_lds(&sStage[wid][0][0], 136, m, 64 + ks * 32, hi8),
                       bfrag(sWV2, 72, m, ks * 32, khi16), av);
      if (m == 0) {  // n==0 lanes hold v_w for edges r+hi8
        float bv = sBV2;
#pragma unroll
        for (int r = 0; r < 8; ++r) sVW[wid][r + hi8] = av[r] + bv;
      }
    }
    __syncthreads();

    // ---- scatter m_v_e = v_w * rel_pos and edge counts ----
    if (active && lane < 16) {
      int d = sDst[wid][lane];
      if (d >= 0) {
        float vw = sVW[wid][lane];
        atomicAdd(&mv[2 * d],     vw * sRelX[wid][lane]);
        atomicAdd(&mv[2 * d + 1], vw * sRelY[wid][lane]);
        atomicAdd(&cnt[d], 1.0f);
      }
    }

    // ---- phi_e layer 2 ----
    v8f acc2[4];
#pragma unroll
    for (int i = 0; i < 4; ++i) acc2[i] = v8f_zero();
    if (active) {
#pragma unroll
      for (int ks = 0; ks < 2; ++ks) {
        v16bf a = afrag_lds(&sStage[wid][0][0], 136, m, ks * 32, hi8);
#pragma unroll
        for (int nt = 0; nt < 4; ++nt)
          acc2[nt] = wmma_bf16(a, bfrag(sWE2, 72, nt * 16 + m, ks * 32, khi16),
                               acc2[nt]);
      }
    }
    __syncthreads();
    if (active) {
#pragma unroll
      for (int nt = 0; nt < 4; ++nt) {
        int n = nt * 16 + m;
        float bias = sBE2[n];
#pragma unroll
        for (int r = 0; r < 8; ++r)
          sStage[wid][r + hi8][n] = f2bf(softplus_f(acc2[nt][r] + bias));
      }
    }
    __syncthreads();

    // ---- phi_e layer 3 + atomic scatter of m_h_e ----
    if (active) {
      v8f acc3[4];
#pragma unroll
      for (int i = 0; i < 4; ++i) acc3[i] = v8f_zero();
#pragma unroll
      for (int ks = 0; ks < 2; ++ks) {
        v16bf a = afrag_lds(&sStage[wid][0][0], 136, m, ks * 32, hi8);
#pragma unroll
        for (int nt = 0; nt < 4; ++nt)
          acc3[nt] = wmma_bf16(a, bfrag(sWE3, 72, nt * 16 + m, ks * 32, khi16),
                               acc3[nt]);
      }
#pragma unroll
      for (int nt = 0; nt < 4; ++nt) {
        int n = nt * 16 + m;
        float bias = sBE3[n];
#pragma unroll
        for (int r = 0; r < 8; ++r) {
          int d = sDst[wid][r + hi8];
          if (d >= 0) atomicAdd(&mh[(size_t)d * 64 + n], acc3[nt][r] + bias);
        }
      }
    }
    __syncthreads();
  }
}

// ---------------- node kernel: mean, norm, phi_h, residual ----------------
__global__ __launch_bounds__(256) void egnn_node_kernel(
    const float* __restrict__ x, const float* __restrict__ mh,
    const float* __restrict__ mv, const float* __restrict__ cnt,
    const unsigned short* __restrict__ gWH1, const unsigned short* __restrict__ gWH2,
    const float* __restrict__ gBias, float* __restrict__ out, int N, int ntiles,
    int niters) {
  __shared__ __attribute__((aligned(16))) unsigned short sWH1[64 * 168];
  __shared__ __attribute__((aligned(16))) unsigned short sWH2[64 * 72];
  __shared__ __attribute__((aligned(16))) unsigned short sStage[NW][16][72];
  __shared__ float sBH1[64], sBH2[64];
  __shared__ float sInv[NW][16], sNorm[NW][16];

  for (int i = threadIdx.x; i < 64 * 168 / 2; i += blockDim.x)
    ((unsigned int*)sWH1)[i] = ((const unsigned int*)gWH1)[i];
  for (int i = threadIdx.x; i < 64 * 72 / 2; i += blockDim.x)
    ((unsigned int*)sWH2)[i] = ((const unsigned int*)gWH2)[i];
  for (int i = threadIdx.x; i < 64; i += blockDim.x) {
    sBH1[i] = gBias[257 + i];
    sBH2[i] = gBias[321 + i];
  }
  __syncthreads();

  const int wid   = threadIdx.x >> 5;
  const int lane  = threadIdx.x & 31;
  const int m     = lane & 15;
  const int hi8   = (lane >> 4) << 3;
  const int khi16 = hi8 << 1;

  for (int it = 0; it < niters; ++it) {
    int tile = (it * gridDim.x + blockIdx.x) * NW + wid;
    bool active = tile < ntiles;

    if (active && lane < 16) {
      int i = tile * 16 + lane;
      if (i >= N) i = N - 1;
      float inv = 1.0f / fmaxf(cnt[i], 1.0f);
      float mvx = mv[2 * i] * inv, mvy = mv[2 * i + 1] * inv;
      sInv[wid][lane]  = inv;
      sNorm[wid][lane] = sqrtf(mvx * mvx + mvy * mvy);
    }
    __syncthreads();

    v8f acc[4];
#pragma unroll
    for (int i = 0; i < 4; ++i) acc[i] = v8f_zero();
    if (active) {
      int i = tile * 16 + m;
      if (i >= N) i = N - 1;
      const float* xr = x  + (size_t)i * 64;
      const float* mr = mh + (size_t)i * 64;
      float inv = sInv[wid][m];
#pragma unroll
      for (int ks = 0; ks < 4; ++ks) {
        v16bf a = (ks < 2) ? afrag_g(xr, ks * 32, hi8, 1.0f)
                           : afrag_g(mr, (ks - 2) * 32, hi8, inv);  // mean on the fly
#pragma unroll
        for (int nt = 0; nt < 4; ++nt)
          acc[nt] = wmma_bf16(a, bfrag(sWH1, 168, nt * 16 + m, ks * 32, khi16),
                              acc[nt]);
      }
      {  // k-step 4: feature 128 = ||m_v||
        Frag fa;
#pragma unroll
        for (int i2 = 0; i2 < 8; ++i2) fa.u[i2] = 0u;
        if (lane < 16) fa.u[0] = pack2(sNorm[wid][m], 0.0f);
#pragma unroll
        for (int nt = 0; nt < 4; ++nt)
          acc[nt] = wmma_bf16(fa.v, bfrag(sWH1, 168, nt * 16 + m, 128, khi16),
                              acc[nt]);
      }
#pragma unroll
      for (int nt = 0; nt < 4; ++nt) {
        int n = nt * 16 + m;
        float bias = sBH1[n];
#pragma unroll
        for (int r = 0; r < 8; ++r)
          sStage[wid][r + hi8][n] = f2bf(softplus_f(acc[nt][r] + bias));
      }
    }
    __syncthreads();

    if (active) {
      v8f acc2[4];
#pragma unroll
      for (int i = 0; i < 4; ++i) acc2[i] = v8f_zero();
#pragma unroll
      for (int ks = 0; ks < 2; ++ks) {
        v16bf a = afrag_lds(&sStage[wid][0][0], 72, m, ks * 32, hi8);
#pragma unroll
        for (int nt = 0; nt < 4; ++nt)
          acc2[nt] = wmma_bf16(a, bfrag(sWH2, 72, nt * 16 + m, ks * 32, khi16),
                               acc2[nt]);
      }
#pragma unroll
      for (int nt = 0; nt < 4; ++nt) {
        int n = nt * 16 + m;
        float bias = sBH2[n];
#pragma unroll
        for (int r = 0; r < 8; ++r) {
          int row = tile * 16 + r + hi8;
          if (row < N)
            out[(size_t)row * 64 + n] = x[(size_t)row * 64 + n] + acc2[nt][r] + bias;
        }
      }
    }
    __syncthreads();
  }
}

// ===========================================================================
extern "C" void kernel_launch(void* const* d_in, const int* in_sizes, int n_in,
                              void* d_out, int out_size, void* d_ws, size_t ws_size,
                              hipStream_t stream) {
  // setup_inputs() order (nested tuples flattened in order):
  const float* x   = (const float*)d_in[0];
  const float* pos = (const float*)d_in[1];
  const float* vel = (const float*)d_in[2];
  const int*   ei  = (const int*)d_in[3];
  const float* We1 = (const float*)d_in[4];   const float* be1 = (const float*)d_in[5];
  const float* We2 = (const float*)d_in[6];   const float* be2 = (const float*)d_in[7];
  const float* We3 = (const float*)d_in[8];   const float* be3 = (const float*)d_in[9];
  const float* Wv1 = (const float*)d_in[10];  const float* bv1 = (const float*)d_in[11];
  const float* Wv2 = (const float*)d_in[12];  const float* bv2 = (const float*)d_in[13];
  const float* Wh1 = (const float*)d_in[14];  const float* bh1 = (const float*)d_in[15];
  const float* Wh2 = (const float*)d_in[16];  const float* bh2 = (const float*)d_in[17];

  const int N = in_sizes[0] / 64;
  const int E = in_sizes[3] / 2;
  const int* srcIdx = ei;
  const int* dstIdx = ei + E;

  // ---- workspace carve (mh, mv, cnt contiguous for one zero pass) ----
  char* ws = (char*)d_ws;
  size_t off = 0;
  float* mh  = (float*)(ws + off); off += (size_t)N * 64 * sizeof(float);
  float* mv  = (float*)(ws + off); off += (size_t)N * 2 * sizeof(float);
  float* cnt = (float*)(ws + off); off += (size_t)N * sizeof(float);
  off = (off + 255) & ~(size_t)255;
  unsigned short* gWE1 = (unsigned short*)(ws + off); off += 128 * 168 * 2;
  unsigned short* gWE2 = (unsigned short*)(ws + off); off += 64 * 72 * 2;
  unsigned short* gWE3 = (unsigned short*)(ws + off); off += 64 * 72 * 2;
  unsigned short* gWV2 = (unsigned short*)(ws + off); off += 16 * 72 * 2;
  unsigned short* gWH1 = (unsigned short*)(ws + off); off += 64 * 168 * 2;
  unsigned short* gWH2 = (unsigned short*)(ws + off); off += 64 * 72 * 2;
  off = (off + 255) & ~(size_t)255;
  float* gBias = (float*)(ws + off); off += 512 * sizeof(float);

  // ---- phase 0: zero accumulators + prep bf16 weights ----
  size_t zn = (size_t)N * 67;  // mh + mv + cnt
  zero_f32<<<4096, 256, 0, stream>>>(mh, zn);
  prep_weights<<<128, 256, 0, stream>>>(We1, Wv1, We2, We3, Wv2, Wh1, Wh2,
                                        be1, bv1, be2, be3, bv2, bh1, bh2,
                                        gWE1, gWE2, gWE3, gWV2, gWH1, gWH2, gBias);

  // ---- phase 1: edges ----
  int ntE = (E + 15) / 16;
  int niterE = (ntE + EBLOCKS * NW - 1) / (EBLOCKS * NW);
  egnn_edge_kernel<<<EBLOCKS, 256, 0, stream>>>(x, pos, vel, srcIdx, dstIdx,
                                                gWE1, gWE2, gWE3, gWV2, gBias,
                                                mh, mv, cnt, E, ntE, niterE);

  // ---- phase 2: nodes ----
  int ntN = (N + 15) / 16;
  int niterN = (ntN + NBLOCKS * NW - 1) / (NBLOCKS * NW);
  egnn_node_kernel<<<NBLOCKS, 256, 0, stream>>>(x, mh, mv, cnt, gWH1, gWH2, gBias,
                                                (float*)d_out, N, ntN, niterN);
}
